// GNOT_13726715478157
// MI455X (gfx1250) — matrix operand order repaired and checked by reading, not compiled, not run on hardware
//
#include <hip/hip_runtime.h>
#include <hip/hip_bf16.h>

typedef __bf16 bf16_t;
typedef __attribute__((ext_vector_type(16))) __bf16 v16bf;
typedef __attribute__((ext_vector_type(8)))  __bf16 v8bf;
typedef __attribute__((ext_vector_type(8)))  float  v8f;
typedef __attribute__((ext_vector_type(4)))  float  v4f;

#define BM 64
#define BN 128
#define BK 32

__device__ __forceinline__ float gelu_exact(float x) {
  return 0.5f * x * (1.0f + erff(x * 0.70710678118654752f));
}

// ---------------------------------------------------------------------------
// Convert fp32 weight W[K][N] (row-major) -> bf16 Wt[Npad][Kpad] (N-major,
// K contiguous) zero-padded, Npad multiple of BN so GEMM B staging is blind.
// ---------------------------------------------------------------------------
__global__ __launch_bounds__(256) void k_convert_w(
    const float* __restrict__ W, bf16_t* __restrict__ Wt,
    int K, int N, int Kpad, int Npad)
{
  int idx = blockIdx.x * 256 + threadIdx.x;
  int total = Kpad * Npad;
  if (idx >= total) return;
  int n = idx / Kpad;
  int k = idx - n * Kpad;
  float v = (k < K && n < N) ? W[(size_t)k * N + n] : 0.0f;
  Wt[idx] = (bf16_t)v;
}

// ---------------------------------------------------------------------------
// bf16 WMMA GEMM: C[M][N] = epi(A[M,K] @ W[K,N] + bias)
// 256 threads = 8 waves; 64x128 C tile; wave (w%4, w/4) -> 16 rows x 64 cols
// = 4 x (16x16) accumulators; K stepped by 32.
// Pipeline: double-buffered LDS, ONE barrier per K-step.
//   - B tile: global_load_async_to_lds_b128 (ASYNCcnt), issued one tile ahead
//   - A tile: fp32 fetched to regs one tile ahead, cvt->bf16 at store time
// Fragments: contiguous k = 16*(lane/16) + s, same for A 16x32 and B 32x16.
// ---------------------------------------------------------------------------
template <bool FULLK>
__global__ __launch_bounds__(256) void k_gemm_wmma(
    const float* __restrict__ A, int lda,
    const bf16_t* __restrict__ Wt,
    const float* __restrict__ bias,
    float* __restrict__ C,
    const float* __restrict__ res,
    const float* __restrict__ scale, int sstride,
    int K, int Kpad, int N,
    int act, int accum)
{
  __shared__ __align__(64) bf16_t As[2][BM][BK];
  __shared__ __align__(64) bf16_t Bs[2][BN][BK];

  const int tid  = threadIdx.x;
  const int lane = tid & 31;
  const int wave = tid >> 5;
  const int m0 = blockIdx.x * BM;
  const int n0 = blockIdx.y * BN;
  const int mw = (wave & 3) * 16;    // wave m offset within tile
  const int nw = (wave >> 2) * 64;   // wave n offset within tile
  const int g    = lane >> 4;        // lane group (0/1)
  const int lrow = lane & 15;

  v8f acc0 = {}, acc1 = {}, acc2 = {}, acc3 = {};

  const int ar = tid >> 2;           // A staging row 0..63
  const int ac = (tid & 3) * 8;      // A staging col 0/8/16/24
  const int br = tid >> 1;           // B staging row 0..127
  const int bc = (tid & 1) * 16;     // B staging col 0/16

  const float*  aptr = A + (size_t)(m0 + ar) * lda + ac;
  const bf16_t* bptr = Wt + (size_t)(n0 + br) * Kpad + bc;
  const int nk = Kpad / BK;

  // ---- prologue: async B tile 0 -> Bs[0]; A tile 0 -> regs ----
  {
    unsigned l0 = (unsigned)(size_t)&Bs[0][br][bc];
    unsigned l1 = (unsigned)(size_t)&Bs[0][br][bc + 8];
    const bf16_t* gp = bptr;
    asm volatile("global_load_async_to_lds_b128 %0, %1, off"
                 :: "v"(l0), "v"(gp) : "memory");
    asm volatile("global_load_async_to_lds_b128 %0, %1, off"
                 :: "v"(l1), "v"(gp + 8) : "memory");
  }
  v4f fa0 = {}, fa1 = {};
  if (FULLK) {
    fa0 = *(const v4f*)aptr;
    fa1 = *(const v4f*)(aptr + 4);
  } else {
    #pragma unroll
    for (int j = 0; j < 4; ++j) {
      fa0[j] = (ac + j     < K) ? aptr[j]     : 0.0f;
      fa1[j] = (ac + j + 4 < K) ? aptr[j + 4] : 0.0f;
    }
  }

  for (int i = 0; i < nk; ++i) {
    const int cur = i & 1;
    const int nxt = cur ^ 1;
    const int k0 = i * BK;

    // ---- convert + store A tile i (regs fetched last iteration) ----
    {
      v8bf t;
      #pragma unroll
      for (int j = 0; j < 4; ++j) { t[j] = (bf16_t)fa0[j]; t[4 + j] = (bf16_t)fa1[j]; }
      *(v8bf*)&As[cur][ar][ac] = t;
    }
    // ---- fetch A tile i+1 into regs (overlaps barrier + WMMA phase) ----
    if (i + 1 < nk) {
      const float* ap = aptr + k0 + BK;
      if (FULLK) {
        fa0 = *(const v4f*)ap;
        fa1 = *(const v4f*)(ap + 4);
      } else {
        #pragma unroll
        for (int j = 0; j < 4; ++j) {
          fa0[j] = (k0 + BK + ac + j     < K) ? ap[j]     : 0.0f;
          fa1[j] = (k0 + BK + ac + j + 4 < K) ? ap[j + 4] : 0.0f;
        }
      }
      __builtin_prefetch(aptr + k0 + 2 * BK, 0, 3);
    }
    // B tile i must be fully in LDS before the barrier releases readers.
    asm volatile("s_wait_asynccnt 0x0" ::: "memory");
    __syncthreads();

    // ---- issue async B tile i+1 -> Bs[nxt] (safe: post-barrier) ----
    if (i + 1 < nk) {
      unsigned l0 = (unsigned)(size_t)&Bs[nxt][br][bc];
      unsigned l1 = (unsigned)(size_t)&Bs[nxt][br][bc + 8];
      const bf16_t* gp = bptr + k0 + BK;
      asm volatile("global_load_async_to_lds_b128 %0, %1, off"
                   :: "v"(l0), "v"(gp) : "memory");
      asm volatile("global_load_async_to_lds_b128 %0, %1, off"
                   :: "v"(l1), "v"(gp + 8) : "memory");
    }

    // ---- fragments + 4 WMMAs ----
    v16bf afrag;
    {
      v8bf lo = *(const v8bf*)&As[cur][mw + lrow][16 * g];
      v8bf hi = *(const v8bf*)&As[cur][mw + lrow][16 * g + 8];
      #pragma unroll
      for (int j = 0; j < 8; ++j) { afrag[j] = lo[j]; afrag[8 + j] = hi[j]; }
    }
    v16bf b0 = *(const v16bf*)&Bs[cur][nw +  0 + lrow][16 * g];
    v16bf b1 = *(const v16bf*)&Bs[cur][nw + 16 + lrow][16 * g];
    v16bf b2 = *(const v16bf*)&Bs[cur][nw + 32 + lrow][16 * g];
    v16bf b3 = *(const v16bf*)&Bs[cur][nw + 48 + lrow][16 * g];

    acc0 = __builtin_amdgcn_wmma_f32_16x16x32_bf16(false, afrag, false, b0,
                                                   (short)0, acc0, false, false);
    acc1 = __builtin_amdgcn_wmma_f32_16x16x32_bf16(false, afrag, false, b1,
                                                   (short)0, acc1, false, false);
    acc2 = __builtin_amdgcn_wmma_f32_16x16x32_bf16(false, afrag, false, b2,
                                                   (short)0, acc2, false, false);
    acc3 = __builtin_amdgcn_wmma_f32_16x16x32_bf16(false, afrag, false, b3,
                                                   (short)0, acc3, false, false);
  }

  // --- epilogue: VGPR r, lane L -> M = 8*(L/16)+r, N = L%16 ---
  const int mbase = m0 + mw + 8 * g;
  #pragma unroll
  for (int r = 0; r < 8; ++r) {
    int m = mbase + r;
    #pragma unroll
    for (int s = 0; s < 4; ++s) {
      int n = n0 + nw + 16 * s + lrow;
      if (n >= N) continue;
      float v = (s == 0) ? acc0[r] : (s == 1) ? acc1[r] : (s == 2) ? acc2[r] : acc3[r];
      if (bias)  v += bias[n];
      if (act)   v = gelu_exact(v);
      if (scale) v *= scale[(size_t)m * sstride];
      if (res)   v += res[(size_t)m * N + n];
      size_t off = (size_t)m * N + n;
      if (accum) C[off] += v; else C[off] = v;
    }
  }
}

// ---------------------------------------------------------------------------
// LayerNorm over D=256, one wave per row. addto=1: out += LN(in)
// ---------------------------------------------------------------------------
__global__ __launch_bounds__(256) void k_layernorm(
    const float* __restrict__ in, const float* __restrict__ g,
    const float* __restrict__ b, float* __restrict__ out, int addto)
{
  int row  = blockIdx.x * 8 + (threadIdx.x >> 5);
  int lane = threadIdx.x & 31;
  const float* p = in + (size_t)row * 256;
  float v[8];
  float s = 0.f;
  #pragma unroll
  for (int i = 0; i < 8; ++i) { v[i] = p[lane + 32 * i]; s += v[i]; }
  #pragma unroll
  for (int o = 16; o > 0; o >>= 1) s += __shfl_xor(s, o, 32);
  float mean = s * 0.00390625f;
  float q = 0.f;
  #pragma unroll
  for (int i = 0; i < 8; ++i) { float d = v[i] - mean; q += d * d; }
  #pragma unroll
  for (int o = 16; o > 0; o >>= 1) q += __shfl_xor(q, o, 32);
  float inv = rsqrtf(q * 0.00390625f + 1e-5f);
  float* op = out + (size_t)row * 256;
  #pragma unroll
  for (int i = 0; i < 8; ++i) {
    int c = lane + 32 * i;
    float y = (v[i] - mean) * inv * g[c] + b[c];
    if (addto) op[c] += y; else op[c] = y;
  }
}

// per-head (32-channel) softmax over [M][256], in place; wave per row
__global__ __launch_bounds__(256) void k_softmax_heads(float* __restrict__ buf)
{
  int row  = blockIdx.x * 8 + (threadIdx.x >> 5);
  int lane = threadIdx.x & 31;
  float* p = buf + (size_t)row * 256;
  #pragma unroll
  for (int h = 0; h < 8; ++h) {
    float v = p[h * 32 + lane];
    float mx = v;
    for (int o = 16; o > 0; o >>= 1) mx = fmaxf(mx, __shfl_xor(mx, o, 32));
    float e = expf(v - mx);
    float s = e;
    for (int o = 16; o > 0; o >>= 1) s += __shfl_xor(s, o, 32);
    p[h * 32 + lane] = e / s;
  }
}

__global__ __launch_bounds__(256) void k_softmax4(float* __restrict__ g, int M)
{
  int m = blockIdx.x * 256 + threadIdx.x;
  if (m >= M) return;
  float* p = g + (size_t)m * 4;
  float a = p[0], b = p[1], c = p[2], d = p[3];
  float mx = fmaxf(fmaxf(a, b), fmaxf(c, d));
  a = expf(a - mx); b = expf(b - mx); c = expf(c - mx); d = expf(d - mx);
  float inv = 1.f / (a + b + c + d);
  p[0] = a * inv; p[1] = b * inv; p[2] = c * inv; p[3] = d * inv;
}

__global__ __launch_bounds__(256) void k_zero(float* __restrict__ p, int n)
{
  int i = blockIdx.x * 256 + threadIdx.x;
  if (i < n) p[i] = 0.f;
}

// ---------------------------------------------------------------------------
// Linear-attention reduction: ctx[bh][d][e] = sum_t k[t][d]*v[t][e],
// ksum[bh][d] = sum_t k[t][d].  grid = (T/256 chunks, B*H), f32 atomics.
// ---------------------------------------------------------------------------
__global__ __launch_bounds__(256) void k_attn_reduce(
    const float* __restrict__ Kb, const float* __restrict__ Vb,
    float* __restrict__ ctx, float* __restrict__ ksum, int T)
{
  __shared__ float ks[256][33];
  __shared__ float vs[256][33];
  int bh = blockIdx.y;
  int b = bh >> 3, h = bh & 7;
  int t0 = blockIdx.x * 256;
  const float* kp = Kb + ((size_t)b * T + t0) * 256 + h * 32;
  const float* vp = Vb + ((size_t)b * T + t0) * 256 + h * 32;
  for (int idx = threadIdx.x; idx < 256 * 32; idx += 256) {
    int r = idx >> 5, c = idx & 31;
    ks[r][c] = kp[(size_t)r * 256 + c];
    vs[r][c] = vp[(size_t)r * 256 + c];
  }
  __syncthreads();
  int d  = threadIdx.x >> 3;
  int e0 = (threadIdx.x * 4) & 31;
  float a0 = 0.f, a1 = 0.f, a2 = 0.f, a3 = 0.f;
  for (int t = 0; t < 256; ++t) {
    float kd = ks[t][d];
    a0 += kd * vs[t][e0 + 0];
    a1 += kd * vs[t][e0 + 1];
    a2 += kd * vs[t][e0 + 2];
    a3 += kd * vs[t][e0 + 3];
  }
  float* cp = ctx + (size_t)bh * 1024 + d * 32 + e0;
  atomicAdd(cp + 0, a0); atomicAdd(cp + 1, a1);
  atomicAdd(cp + 2, a2); atomicAdd(cp + 3, a3);
  if (threadIdx.x < 32) {
    float s = 0.f;
    for (int t = 0; t < 256; ++t) s += ks[t][threadIdx.x];
    atomicAdd(ksum + (size_t)bh * 32 + threadIdx.x, s);
  }
}

// out[t] = q[t] + (q[t] @ ctx) / (q[t] . ksum); block = token, wave = head
__global__ __launch_bounds__(256) void k_attn_apply(
    const float* __restrict__ Q, const float* __restrict__ ctx,
    const float* __restrict__ ksum, float* __restrict__ out, int T)
{
  int m = blockIdx.x;
  int b = m / T;
  int h = threadIdx.x >> 5;
  int lane = threadIdx.x & 31;
  int bh = b * 8 + h;
  float qv = Q[(size_t)m * 256 + h * 32 + lane];
  float ds = qv * ksum[(size_t)bh * 32 + lane];
  for (int o = 16; o > 0; o >>= 1) ds += __shfl_xor(ds, o, 32);
  float dinv = 1.0f / ds;
  const float* cx = ctx + (size_t)bh * 1024;
  float y = 0.f;
  #pragma unroll
  for (int dd = 0; dd < 32; ++dd) {
    float qd = __shfl(qv, dd, 32);
    y += qd * cx[dd * 32 + lane];
  }
  out[(size_t)m * 256 + h * 32 + lane] = qv + y * dinv;
}

// ===========================================================================
extern "C" void kernel_launch(void* const* d_in, const int* in_sizes, int n_in,
                              void* d_out, int out_size, void* d_ws, size_t ws_size,
                              hipStream_t stream)
{
  (void)in_sizes; (void)out_size; (void)ws_size;
  if (n_in < 218) return;

  const int Mtok = 4 * 8192;  // B*T
  const int T = 8192;
  auto in = [&](int i) { return (const float*)d_in[i]; };

  // ---- workspace layout (fp32 buffers, then bf16 weight pool) ----
  float* fws = (float*)d_ws;
  size_t off = 0;
  auto alloc = [&](size_t n) { float* p = fws + off; off += n; return p; };
  float* xb   = alloc((size_t)Mtok * 256);
  float* zb   = alloc((size_t)Mtok * 256);
  float* ynb  = alloc((size_t)Mtok * 256);
  float* xnb  = alloc((size_t)Mtok * 256);
  float* qb   = alloc((size_t)Mtok * 256);   // also MLP temp 0
  float* kb   = alloc((size_t)Mtok * 256);   // also MLP temp 1
  float* vb   = alloc((size_t)Mtok * 256);
  float* ab   = alloc((size_t)Mtok * 256);   // attn out / moe accumulator
  float* gbuf = alloc((size_t)Mtok * 4);
  float* h0   = alloc((size_t)Mtok * 1024);
  float* h1   = alloc((size_t)Mtok * 1024);
  float* ctxb = alloc(32 * 1024);
  float* ksb  = alloc(32 * 32);
  off = (off + 63) & ~(size_t)63;
  bf16_t* wpool = (bf16_t*)(fws + off);
  size_t wo = 0;

  auto gemm = [&](const float* A, int lda, const float* W, const float* bias,
                  float* C, int K, int N, int act,
                  const float* res, const float* scale, int sstride, int accum) {
    int Kpad = (K + 31) & ~31;
    int Npad = (N + BN - 1) & ~(BN - 1);      // pad to block tile -> blind B reads
    bf16_t* Wt = wpool + wo;
    wo += (size_t)Kpad * Npad;
    int tot = Kpad * Npad;
    k_convert_w<<<dim3((tot + 255) / 256), dim3(256), 0, stream>>>(W, Wt, K, N, Kpad, Npad);
    dim3 grid(Mtok / BM, Npad / BN);
    if (K == Kpad) {
      k_gemm_wmma<true><<<grid, dim3(256), 0, stream>>>(A, lda, Wt, bias, C, res,
                                                        scale, sstride, K, Kpad, N,
                                                        act, accum);
    } else {
      k_gemm_wmma<false><<<grid, dim3(256), 0, stream>>>(A, lda, Wt, bias, C, res,
                                                         scale, sstride, K, Kpad, N,
                                                         act, accum);
    }
  };
  auto ln = [&](const float* src, const float* g, const float* b, float* dst, int addto) {
    k_layernorm<<<dim3(Mtok / 8), dim3(256), 0, stream>>>(src, g, b, dst, addto);
  };
  auto smheads = [&](float* buf) {
    k_softmax_heads<<<dim3(Mtok / 8), dim3(256), 0, stream>>>(buf);
  };

  // ---- parameter pointers, setup_inputs() insertion-order flattening ----
  int ix = 2;
  const float *tw[4], *tb[4];
  for (int i = 0; i < 4; ++i) { tw[i] = in(ix++); tb[i] = in(ix++); }
  const float *bw[4], *bbv[4];
  for (int i = 0; i < 4; ++i) { bw[i] = in(ix++); bbv[i] = in(ix++); }

  struct BP {
    const float *ln1g, *ln1b, *ln2g, *ln2b, *ln3g, *ln3b, *ln4g, *ln4b, *ln5g, *ln5b;
    const float *saw[4], *sab[4];      // q,k,v,proj
    const float *caw[4], *cab[4];      // q,keys0,values0,proj
    const float *m1w1[4], *m1b1[4], *m1w2[4], *m1b2[4];
    const float *m2w1[4], *m2b1[4], *m2w2[4], *m2b2[4];
    const float *gw[3], *gbp[3];
  } bp[3];
  for (int l = 0; l < 3; ++l) {
    BP& p = bp[l];
    p.ln1g = in(ix++); p.ln1b = in(ix++);
    p.ln2g = in(ix++); p.ln2b = in(ix++);
    p.ln3g = in(ix++); p.ln3b = in(ix++);
    p.ln4g = in(ix++); p.ln4b = in(ix++);
    p.ln5g = in(ix++); p.ln5b = in(ix++);
    for (int i = 0; i < 4; ++i) { p.saw[i] = in(ix++); p.sab[i] = in(ix++); }
    for (int i = 0; i < 4; ++i) { p.caw[i] = in(ix++); p.cab[i] = in(ix++); }
    for (int e = 0; e < 4; ++e) {
      p.m1w1[e] = in(ix++); p.m1b1[e] = in(ix++);
      p.m1w2[e] = in(ix++); p.m1b2[e] = in(ix++);
    }
    for (int e = 0; e < 4; ++e) {
      p.m2w1[e] = in(ix++); p.m2b1[e] = in(ix++);
      p.m2w2[e] = in(ix++); p.m2b2[e] = in(ix++);
    }
    for (int i = 0; i < 3; ++i) { p.gw[i] = in(ix++); p.gbp[i] = in(ix++); }
  }
  const float *ow[4], *ob[4];
  for (int i = 0; i < 4; ++i) { ow[i] = in(ix++); ob[i] = in(ix++); }

  const float* xin  = in(0);  // (B,T,2) -- also 'pos' since SPACE==TRUNK
  const float* inp0 = in(1);  // (B,T,3)

  auto mlp = [&](const float* X, int din, const float* const* W,
                 const float* const* Bv, float* out, int dout) {
    gemm(X, din, W[0], Bv[0], qb, din, 256, 1, nullptr, nullptr, 0, 0);
    gemm(qb, 256, W[1], Bv[1], kb, 256, 256, 1, qb, nullptr, 0, 0);
    gemm(kb, 256, W[2], Bv[2], qb, 256, 256, 1, kb, nullptr, 0, 0);
    gemm(qb, 256, W[3], Bv[3], out, 256, dout, 0, nullptr, nullptr, 0, 0);
  };
  auto attention = [&](const float* qsrc, const float* kvsrc,
                       const float* wq, const float* bq,
                       const float* wk, const float* bk,
                       const float* wv, const float* bv2,
                       const float* wp, const float* bp2) {
    gemm(qsrc, 256, wq, bq, qb, 256, 256, 0, nullptr, nullptr, 0, 0);
    smheads(qb);
    gemm(kvsrc, 256, wk, bk, kb, 256, 256, 0, nullptr, nullptr, 0, 0);
    smheads(kb);
    gemm(kvsrc, 256, wv, bv2, vb, 256, 256, 0, nullptr, nullptr, 0, 0);
    k_zero<<<dim3((32 * 1024 + 1024 + 255) / 256), dim3(256), 0, stream>>>(ctxb, 32 * 1024 + 1024);
    k_attn_reduce<<<dim3(T / 256, 32), dim3(256), 0, stream>>>(kb, vb, ctxb, ksb, T);
    k_attn_apply<<<dim3(Mtok), dim3(256), 0, stream>>>(qb, ctxb, ksb, ab, T);
    gemm(ab, 256, wp, bp2, xb, 256, 256, 0, xb, nullptr, 0, 0);  // x += proj(attn)
  };
  auto moe = [&](const float* const* w1, const float* const* b1,
                 const float* const* w2, const float* const* b2,
                 const float* lng, const float* lnb) {
    for (int e = 0; e < 4; ++e) {
      gemm(xb, 256, w1[e], b1[e], h0, 256, 1024, 1, nullptr, nullptr, 0, 0);
      gemm(h0, 1024, w2[e], b2[e], ab, 1024, 256, 0, nullptr, gbuf + e, 4, e > 0);
    }
    ln(ab, lng, lnb, xb, 1);  // x += LN(moe_out)
  };

  // ---- forward pass ----
  mlp(xin, 2, tw, tb, xb, 256);    // trunk
  mlp(inp0, 3, bw, bbv, zb, 256);  // branch[0]

  for (int l = 0; l < 3; ++l) {
    BP& p = bp[l];
    // gate(pos): 2 -> 1024 -> 1024 -> 4 -> softmax
    gemm(xin, 2, p.gw[0], p.gbp[0], h0, 2, 1024, 1, nullptr, nullptr, 0, 0);
    gemm(h0, 1024, p.gw[1], p.gbp[1], h1, 1024, 1024, 1, nullptr, nullptr, 0, 0);
    gemm(h1, 1024, p.gw[2], p.gbp[2], gbuf, 1024, 4, 0, nullptr, nullptr, 0, 0);
    k_softmax4<<<dim3((Mtok + 255) / 256), dim3(256), 0, stream>>>(gbuf, Mtok);

    ln(zb, p.ln2g, p.ln2b, ynb, 0);  // yn
    ln(xb, p.ln1g, p.ln1b, xnb, 0);  // xn
    attention(xnb, ynb, p.caw[0], p.cab[0], p.caw[1], p.cab[1],
              p.caw[2], p.cab[2], p.caw[3], p.cab[3]);           // cross-attn
    moe(p.m1w1, p.m1b1, p.m1w2, p.m1b2, p.ln3g, p.ln3b);         // moe1
    ln(xb, p.ln4g, p.ln4b, xnb, 0);
    attention(xnb, xnb, p.saw[0], p.sab[0], p.saw[1], p.sab[1],
              p.saw[2], p.sab[2], p.saw[3], p.sab[3]);           // self-attn
    moe(p.m2w1, p.m2b1, p.m2w2, p.m2b2, p.ln5g, p.ln5b);         // moe2
  }

  mlp(xb, 256, ow, ob, (float*)d_out, 3);  // out head (N=3)
}
